// Model_62380105008083
// MI455X (gfx1250) — compile-verified, matrix operand order checked by reading, not run on hardware
//
#include <hip/hip_runtime.h>
#include <hip/hip_bf16.h>
#include <math.h>

typedef _Float16 h16;
typedef __attribute__((ext_vector_type(16))) _Float16 v16h;
typedef __attribute__((ext_vector_type(8)))  _Float16 v8h;
typedef __attribute__((ext_vector_type(8)))  float    v8f;

#define L_LEVELS 10
#define DIM_F    4
#define HIDDEN   64
#define KPAD     64
#define ROWS_PER_BLOCK 128
#define EPS_F 1.1920928955078125e-07f
#define SELU_SCALE 1.0507009873554805f
#define SELU_ALPHA 1.6732632423543772f
#define PI_F 3.14159265358979323846f

__device__ __forceinline__ float softplusf(float x) {
    return x > 20.f ? x : log1pf(expf(x));
}

// Load one 16x32 f16 WMMA A/B fragment from a 64-half contiguous row/column in LDS.
// Per ISA layout: lane<16 -> K = {kc*32+0..7, kc*32+16..23}; lane>=16 -> +8 offset.
__device__ __forceinline__ v16h load_frag64(const h16* base64, int kc, int lane) {
    int ko = kc * 32 + ((lane >> 4) ? 8 : 0);
    v8h lo = *(const v8h*)(base64 + ko);
    v8h hi = *(const v8h*)(base64 + ko + 16);
    v16h r;
#pragma unroll
    for (int i = 0; i < 8; ++i) { r[i] = lo[i]; r[i + 8] = hi[i]; }
    return r;
}

// One M=16 row-tile GEMM: act (16x64 f16, row-major) x W (64xN f16, column-major 64/col)
template <int NTILES>
__device__ __forceinline__ void gemm_tile(const h16* actBase, const h16* wcol,
                                          int lane, v8f* acc) {
    const h16* arow = actBase + (lane & 15) * KPAD;
#pragma unroll
    for (int kc = 0; kc < 2; ++kc) {
        v16h a = load_frag64(arow, kc, lane);
#pragma unroll
        for (int nt = 0; nt < NTILES; ++nt) {
            v16h b = load_frag64(wcol + (nt * 16 + (lane & 15)) * KPAD, kc, lane);
            acc[nt] = __builtin_amdgcn_wmma_f32_16x16x32_f16(
                false, a, false, b, (short)0, acc[nt], false, false);
        }
    }
}

__global__ void __launch_bounds__(256)
tet_hashmlp_kernel(const float* __restrict__ vertices,
                   const int*   __restrict__ indices,
                   const float* __restrict__ camera_center,
                   const float* __restrict__ tables,
                   const float* __restrict__ W1, const float* __restrict__ b1,
                   const float* __restrict__ W2, const float* __restrict__ b2,
                   const float* __restrict__ W3, const float* __restrict__ b3,
                   const float* __restrict__ center,
                   const float* __restrict__ scene_scaling,
                   float* __restrict__ out, int T) {
    __shared__ __align__(16) h16 sW1[HIDDEN * KPAD];        // col-major [n][k]
    __shared__ __align__(16) h16 sW2[HIDDEN * KPAD];
    __shared__ __align__(16) h16 sW3[16 * KPAD];
    __shared__ __align__(16) h16 sAct[8][16 * KPAD];        // per-wave 16-row staging
    __shared__ __align__(16) float sOut[8][16 * 16];        // per-wave final GEMM out
    __shared__ float sVD[ROWS_PER_BLOCK][3];

    const int tid  = threadIdx.x;
    const int lane = tid & 31;
    const int wave = tid >> 5;

    // ---------------- Phase 0: weights -> f16 column-major LDS ----------------
#pragma unroll 4
    for (int i = tid; i < HIDDEN * KPAD; i += 256) {
        int n = i >> 6, k = i & 63;
        sW1[i] = (k < L_LEVELS * DIM_F) ? (h16)W1[k * HIDDEN + n] : (h16)0.f;
        sW2[i] = (h16)W2[k * HIDDEN + n];
    }
    for (int i = tid; i < 16 * KPAD; i += 256) {
        int n = i >> 6, k = i & 63;
        sW3[i] = (h16)W3[k * 16 + n];
    }

    // ---------------- Phase 1: geometry + hash-grid encoding ----------------
    {
        const int r    = tid & (ROWS_PER_BLOCK - 1);   // row within block
        const int half = tid >> 7;                     // 0: levels 0-4, 1: levels 5-9
        int t  = blockIdx.x * ROWS_PER_BLOCK + r;
        int tc = t < T ? t : (T - 1);

        int4 id = ((const int4*)indices)[tc];
        const float* p;
        p = vertices + (size_t)id.x * 3; float v0x = p[0], v0y = p[1], v0z = p[2];
        p = vertices + (size_t)id.y * 3; float v1x = p[0], v1y = p[1], v1z = p[2];
        p = vertices + (size_t)id.z * 3; float v2x = p[0], v2y = p[1], v2z = p[2];
        p = vertices + (size_t)id.w * 3; float v3x = p[0], v3y = p[1], v3z = p[2];

        // A = 2*(vi - v0) + 1e-6*I ; rhs_i = sum(vi^2 - v0^2)
        float a00 = 2.f*(v1x-v0x)+1e-6f, a01 = 2.f*(v1y-v0y),        a02 = 2.f*(v1z-v0z);
        float a10 = 2.f*(v2x-v0x),       a11 = 2.f*(v2y-v0y)+1e-6f,  a12 = 2.f*(v2z-v0z);
        float a20 = 2.f*(v3x-v0x),       a21 = 2.f*(v3y-v0y),        a22 = 2.f*(v3z-v0z)+1e-6f;
        float r0 = (v1x*v1x-v0x*v0x)+(v1y*v1y-v0y*v0y)+(v1z*v1z-v0z*v0z);
        float r1 = (v2x*v2x-v0x*v0x)+(v2y*v2y-v0y*v0y)+(v2z*v2z-v0z*v0z);
        float r2 = (v3x*v3x-v0x*v0x)+(v3y*v3y-v0y*v0y)+(v3z*v3z-v0z*v0z);

        float c00 = a11*a22 - a12*a21, c01 = a10*a22 - a12*a20, c02 = a10*a21 - a11*a20;
        float det = a00*c00 - a01*c01 + a02*c02;
        float inv = 1.f / det;
        float ccx = (r0*c00 - a01*(r1*a22 - a12*r2) + a02*(r1*a21 - a11*r2)) * inv;
        float ccy = (a00*(r1*a22 - a12*r2) - r0*c01 + a02*(a10*r2 - r1*a20)) * inv;
        float ccz = (a00*(a11*r2 - r1*a21) - a01*(a10*r2 - r1*a20) + r0*c02) * inv;

        float radius = sqrtf((ccx-v0x)*(ccx-v0x)+(ccy-v0y)*(ccy-v0y)+(ccz-v0z)*(ccz-v0z));
        float ss = scene_scaling[0];
        float nx = (ccx - center[0]) / ss, ny = (ccy - center[1]) / ss, nz = (ccz - center[2]) / ss;
        float nrm = fmaxf(sqrtf(nx*nx + ny*ny + nz*nz), EPS_F);
        float cvx, cvy, cvz, crf;
        if (nrm <= 1.f) { cvx = nx; cvy = ny; cvz = nz; crf = 1.f; }
        else { float s2 = (2.f - 1.f/nrm) / nrm; cvx = nx*s2; cvy = ny*s2; cvz = nz*s2; crf = 1.f/(nrm*nrm); }
        float cr = radius / ss * crf * 0.5f;
        float xx = cvx * 0.25f + 0.5f, xy = cvy * 0.25f + 0.5f, xz = cvz * 0.25f + 0.5f;

        h16* rowbase = &sAct[r >> 4][(r & 15) * KPAD];
#pragma unroll
        for (int li = 0; li < 5; ++li) {
            int l = half * 5 + li;
            float sc = erff(rsqrtf(fmaxf(8.f * (float)l * cr, EPS_F)));
            float res = (float)(16 << l);
            float px = xx*res, py = xy*res, pz = xz*res;
            float fx = floorf(px), fy = floorf(py), fz = floorf(pz);
            float wx = px - fx, wy = py - fy, wz = pz - fz;
            int bx = (int)fx, by = (int)fy, bz = (int)fz;
            const float4* tl = reinterpret_cast<const float4*>(tables) + ((size_t)l << 16);
            float f0 = 0.f, f1 = 0.f, f2 = 0.f, f3 = 0.f;
#pragma unroll
            for (int c = 0; c < 8; ++c) {
                int ci = (c >> 2) & 1, cj = (c >> 1) & 1, ck = c & 1;
                unsigned hx = (unsigned)(bx + ci);
                unsigned hy = (unsigned)(by + cj) * 2654435761u;
                unsigned hz = (unsigned)(bz + ck) * 805459861u;
                float4 cf = tl[(hx ^ hy ^ hz) & 0xFFFFu];
                float wt = (ci ? wx : 1.f - wx) * (cj ? wy : 1.f - wy) * (ck ? wz : 1.f - wz);
                f0 += cf.x * wt; f1 += cf.y * wt; f2 += cf.z * wt; f3 += cf.w * wt;
            }
            // enc column index = d*L + l (reshape order of stack(...,axis=-1))
            rowbase[0 * L_LEVELS + l] = (h16)(f0 * sc);
            rowbase[1 * L_LEVELS + l] = (h16)(f1 * sc);
            rowbase[2 * L_LEVELS + l] = (h16)(f2 * sc);
            rowbase[3 * L_LEVELS + l] = (h16)(f3 * sc);
        }

        if (half == 0) {
#pragma unroll
            for (int k = L_LEVELS * DIM_F; k < KPAD; ++k) rowbase[k] = (h16)0.f;
            float bcx = (v0x+v1x+v2x+v3x)*0.25f;
            float bcy = (v0y+v1y+v2y+v3y)*0.25f;
            float bcz = (v0z+v1z+v2z+v3z)*0.25f;
            float vdx = camera_center[0]-bcx, vdy = camera_center[1]-bcy, vdz = camera_center[2]-bcz;
            float vn = 1.f / fmaxf(sqrtf(vdx*vdx+vdy*vdy+vdz*vdz), EPS_F);
            sVD[r][0] = vdx*vn; sVD[r][1] = vdy*vn; sVD[r][2] = vdz*vn;
        }
    }
    __syncthreads();
    // From here on, sAct[wave]/sOut[wave] are touched only by their owning wave;
    // same-wave LDS ops are in-order, so no further barriers are required.

    // ---------------- Phase 2: fused MLP via WMMA ----------------
    h16* stg = sAct[wave];
    const int r0  = (lane < 16) ? 0 : 8;
    const int cb  = lane & 15;

    // GEMM1: 16x64 = (16x40 pad 64) @ (64x64), SELU
    {
        v8f acc[4];
#pragma unroll
        for (int nt = 0; nt < 4; ++nt)
#pragma unroll
            for (int j = 0; j < 8; ++j) acc[nt][j] = 0.f;
        gemm_tile<4>(stg, sW1, lane, acc);
#pragma unroll
        for (int nt = 0; nt < 4; ++nt) {
            int col = nt * 16 + cb;
            float bb = b1[col];
#pragma unroll
            for (int j = 0; j < 8; ++j) {
                float x = acc[nt][j] + bb;
                float s = SELU_SCALE * (x > 0.f ? x : SELU_ALPHA * (expf(x) - 1.f));
                stg[(r0 + j) * KPAD + col] = (h16)s;
            }
        }
    }
    // GEMM2: 16x64 @ 64x64, SELU
    {
        v8f acc[4];
#pragma unroll
        for (int nt = 0; nt < 4; ++nt)
#pragma unroll
            for (int j = 0; j < 8; ++j) acc[nt][j] = 0.f;
        gemm_tile<4>(stg, sW2, lane, acc);
#pragma unroll
        for (int nt = 0; nt < 4; ++nt) {
            int col = nt * 16 + cb;
            float bb = b2[col];
#pragma unroll
            for (int j = 0; j < 8; ++j) {
                float x = acc[nt][j] + bb;
                float s = SELU_SCALE * (x > 0.f ? x : SELU_ALPHA * (expf(x) - 1.f));
                stg[(r0 + j) * KPAD + col] = (h16)s;
            }
        }
    }
    // GEMM3: 16x16 = 16x64 @ 64x16
    {
        v8f acc;
#pragma unroll
        for (int j = 0; j < 8; ++j) acc[j] = 0.f;
        gemm_tile<1>(stg, sW3, lane, &acc);
        float bb = b3[cb];
#pragma unroll
        for (int j = 0; j < 8; ++j) sOut[wave][(r0 + j) * 16 + cb] = acc[j] + bb;
    }

    // ---------------- Phase 3: shading (lanes 0..15 of each wave) ----------------
    if (lane < 16) {
        int rr = lane;
        int r  = wave * 16 + rr;
        int t  = blockIdx.x * ROWS_PER_BLOCK + r;
        float o[16];
#pragma unroll
        for (int k = 0; k < 16; ++k) o[k] = sOut[wave][rr * 16 + k];

        float res0 = softplusf(o[0]);
        float res1 = softplusf(o[1]);
        float res2 = softplusf(o[2]);

        float vdx = sVD[r][0], vdy = sVD[r][1], vdz = sVD[r][2];
#pragma unroll
        for (int li = 0; li < 2; ++li) {
            const float* v = &o[4 + li * 6];
            float lc0 = softplusf(v[0] - 3.0f);
            float lc1 = softplusf(v[1] - 3.0f);
            float lc2 = softplusf(v[2] - 3.0f);
            float rough = 4.f * fminf(expf(fminf(fmaxf(v[3] - 1.f, -30.f), 15.f)), 100.f);
            float rd0 = v[4] + (li == 0 ? 0.f : PI_F);
            float rd1 = v[5];
            float srd1 = sinf(rd1);
            float rfx = cosf(rd0) * srd1;
            float rfy = sinf(rd0) * srd1;
            float rfz = cosf(rd1);
            float sim = rfx * vdx + rfy * vdy + rfz * vdz;
            float spec = (sim > 0.f) ? powf(fmaxf(sim, EPS_F), rough) : 0.f;
            res0 += lc0 * spec;
            res1 += lc1 * spec;
            res2 += lc2 * spec;
        }
        if (t < T) {
            out[(size_t)t * 3 + 0] = res0;
            out[(size_t)t * 3 + 1] = res1;
            out[(size_t)t * 3 + 2] = res2;
        }
    }
}

extern "C" void kernel_launch(void* const* d_in, const int* in_sizes, int n_in,
                              void* d_out, int out_size, void* d_ws, size_t ws_size,
                              hipStream_t stream) {
    const float* vertices       = (const float*)d_in[0];
    const int*   indices        = (const int*)  d_in[1];
    const float* camera_center  = (const float*)d_in[2];
    const float* tables         = (const float*)d_in[3];
    const float* W1             = (const float*)d_in[4];
    const float* b1             = (const float*)d_in[5];
    const float* W2             = (const float*)d_in[6];
    const float* b2             = (const float*)d_in[7];
    const float* W3             = (const float*)d_in[8];
    const float* b3             = (const float*)d_in[9];
    const float* center         = (const float*)d_in[10];
    const float* scene_scaling  = (const float*)d_in[11];
    float* out = (float*)d_out;

    int T = in_sizes[1] / 4;
    dim3 grid((T + ROWS_PER_BLOCK - 1) / ROWS_PER_BLOCK);
    dim3 block(256);
    hipLaunchKernelGGL(tet_hashmlp_kernel, grid, block, 0, stream,
                       vertices, indices, camera_center, tables,
                       W1, b1, W2, b2, W3, b3, center, scene_scaling, out, T);
}
